// FeatureGroupTRM_41480794145087
// MI455X (gfx1250) — matrix operand order repaired and checked by reading, not compiled, not run on hardware
//
#include <hip/hip_runtime.h>
#include <hip/hip_bf16.h>
#include <math.h>

typedef _Float16 h16;
typedef __attribute__((ext_vector_type(16))) _Float16 v16h;
typedef __attribute__((ext_vector_type(8)))  _Float16 v8h;
typedef __attribute__((ext_vector_type(8)))  float    v8f;

namespace trm {

constexpr int Bsz = 8192, HB = 16, T = 22, STEPS = 16, XW = 22 * 6 + 200;

// ---------------- LDS layout (bytes) ----------------
constexpr int O_SAIN  = 0;                       // sa_in_w   96x32 f16
constexpr int O_SAOUT = O_SAIN  + 96 * 32 * 2;   // sa_out_w  32x32
constexpr int O_FF1   = O_SAOUT + 32 * 32 * 2;   // sa_ff_w1  64x32
constexpr int O_FF2   = O_FF1   + 64 * 32 * 2;   // sa_ff_w2  32x64
constexpr int O_CE0Q  = O_FF2   + 32 * 64 * 2;   // ce0_in_w rows 0..31
constexpr int O_CE0O  = O_CE0Q  + 32 * 32 * 2;   // ce0_out_w 32x32
constexpr int O_CETIN = O_CE0O  + 32 * 32 * 2;   // cet_in_w  96x32
constexpr int O_CETO  = O_CETIN + 96 * 32 * 2;   // cet_out_w 32x32
constexpr int O_ZFF1  = O_CETO  + 32 * 32 * 2;   // zff_w1    64x96
constexpr int O_ZFF2  = O_ZFF1  + 64 * 96 * 2;   // zff_w2    32x64
constexpr int O_YFF1  = O_ZFF2  + 32 * 64 * 2;   // yff_w1    64x64
constexpr int O_YFF2  = O_YFF1  + 64 * 64 * 2;   // yff_w2    32x64
constexpr int O_ET    = O_YFF2  + 32 * 64 * 2;   // Et  HB x 32 x 32 f16 (rows>=22 stay 0)
constexpr int O_K0    = O_ET  + HB * 32 * 32 * 2;// ce0 keys   HB x 32 x 32 f16
constexpr int O_V0    = O_K0  + HB * 32 * 32 * 2;// ce0 values HB x 32 x 32 f16
constexpr int O_ZH    = O_V0  + HB * 32 * 32 * 2;// z (f16 copy) HB x 32
constexpr int O_Z     = O_ZH  + HB * 32 * 2;     // z f32
constexpr int O_Y     = O_Z   + HB * 32 * 4;     // y f32
constexpr int O_Q0    = O_Y   + HB * 32 * 4;     // ce0 q f16 HB x 32
constexpr int O_QT    = O_Q0  + HB * 32 * 2;     // cet q f16
constexpr int O_AO0   = O_QT  + HB * 32 * 2;     // ce0 attn out f16
constexpr int O_AOT   = O_AO0 + HB * 32 * 2;     // cet attn out f16
constexpr int O_ZCAT  = O_AOT + HB * 32 * 2;     // f16 HB x 96
constexpr int O_YCAT  = O_ZCAT+ HB * 96 * 2;     // f16 HB x 64
constexpr int O_BH    = O_YCAT+ HB * 64 * 2;     // f32 HB x 64 (also BQ0|BQT; prologue kv-weight temp)
constexpr int O_BH16  = O_BH  + HB * 64 * 4;     // f16 HB x 64
constexpr int O_PW    = O_BH16+ HB * 64 * 2;     // per-wave scratch base
constexpr int PW_SCF  = 0;                       // f32 32x96
constexpr int PW_ATT  = 32 * 96 * 4;             // f16 union, 8192 halfs
constexpr int PW_SZ   = PW_ATT + 8192 * 2;       // 28672 B per wave
constexpr int LDS_BYTES = O_PW + 4 * PW_SZ;      // ~288 KB (< 320 KB WGP LDS)

struct TrmParams {
  const float *x;
  const float *tp_w, *tp_b, *tp_g, *tp_be;
  const float *zi_w, *zi_b, *zi_g, *zi_be;
  const float *sa_in_w, *sa_in_b, *sa_out_w, *sa_out_b, *sa_norm_g, *sa_norm_b;
  const float *sa_ff_w1, *sa_ff_b1, *sa_ff_w2, *sa_ff_b2, *sa_ffn_g, *sa_ffn_b;
  const float *ce0_in_w, *ce0_in_b, *ce0_out_w, *ce0_out_b, *ce0_norm_g, *ce0_norm_b;
  const float *cet_in_w, *cet_in_b, *cet_out_w, *cet_out_b, *cet_norm_g, *cet_norm_b;
  const float *zff_w1, *zff_b1, *zff_w2, *zff_b2, *zff_g, *zff_be;
  const float *yff_w1, *yff_b1, *yff_w2, *yff_b2, *yff_g, *yff_be;
  const float *head_w, *head_b;
  float *out;
};

// ---------- WMMA fragment helpers (ISA 7.12.2 layouts, wave32) ----------
__device__ __forceinline__ v16h load_fragA(const h16 *base, int ld, int lane) {
  const int row = lane & 15;
  const int kc  = (lane >> 4) << 3;                  // K chunk 0 or 8
  v8h a = *(const v8h *)(base + row * ld + kc);      // K kc..kc+7
  v8h b = *(const v8h *)(base + row * ld + 16 + kc); // K 16+kc..
  v16h r;
#pragma unroll
  for (int i = 0; i < 8; ++i) { r[i] = a[i]; r[i + 8] = b[i]; }
  return r;
}
// B = W^T: frag[i] = W[n][kbase+kc+i], frag[8+i] = W[n][kbase+16+kc+i] (kc = 8*(lane/16)); row-major weight rows load directly.
__device__ __forceinline__ v16h load_fragB(const h16 *base, int ld, int lane) {
  const h16 *p = base + (lane & 15) * ld + ((lane >> 4) << 3);
  v8h a = *(const v8h *)(p);
  v8h b = *(const v8h *)(p + 16);
  v16h r;
#pragma unroll
  for (int i = 0; i < 8; ++i) { r[i] = a[i]; r[i + 8] = b[i]; }
  return r;
}
__device__ __forceinline__ void store_cfrag(float *base, int ld, int lane, v8f c) {
  const int n = lane & 15, m0 = (lane >> 4) << 3;
#pragma unroll
  for (int v = 0; v < 8; ++v) base[(m0 + v) * ld + n] = c[v];
}

// out[M,N] = A[M,K] @ W[N,K]^T  (16x16x32 tiles, f32 accum to LDS).
// A-frags register-cached across all N-tiles; B-frags reused across M-tiles,
// so LDS frag traffic is MT*KC + NT*KC instead of 2*MT*NT*KC.
template <int MT, int NT, int KC>
__device__ __forceinline__ void mm_lds(const h16 *A, int lda, const h16 *W, int ldw,
                                       float *C, int ldc, int lane) {
  v16h afr[MT * KC];
#pragma unroll
  for (int mt = 0; mt < MT; ++mt)
#pragma unroll
    for (int k = 0; k < KC; ++k)
      afr[mt * KC + k] = load_fragA(A + mt * 16 * lda + 32 * k, lda, lane);
#pragma unroll
  for (int nt = 0; nt < NT; ++nt) {
    v16h bfr[KC];
#pragma unroll
    for (int k = 0; k < KC; ++k)
      bfr[k] = load_fragB(W + nt * 16 * ldw + 32 * k, ldw, lane);
#pragma unroll
    for (int mt = 0; mt < MT; ++mt) {
      v8f c = {0.f, 0.f, 0.f, 0.f, 0.f, 0.f, 0.f, 0.f};
#pragma unroll
      for (int k = 0; k < KC; ++k)
        c = __builtin_amdgcn_wmma_f32_16x16x32_f16(false, afr[mt * KC + k], false,
                                                   bfr[k], (short)0, c, false, false);
      store_cfrag(C + mt * 16 * ldc + nt * 16, ldc, lane, c);
    }
  }
  __builtin_amdgcn_wave_barrier();
}

__device__ __forceinline__ float geluf(float x) {
  return 0.5f * x * (1.0f + erff(x * 0.70710678118654752f));
}
__device__ __forceinline__ void ln32(float *v, const float *g, const float *be) {
  float m = 0.f;
#pragma unroll
  for (int j = 0; j < 32; ++j) m += v[j];
  m *= (1.f / 32.f);
  float var = 0.f;
#pragma unroll
  for (int j = 0; j < 32; ++j) { float d = v[j] - m; var += d * d; }
  var *= (1.f / 32.f);
  float inv = rsqrtf(var + 1e-5f);
#pragma unroll
  for (int j = 0; j < 32; ++j) v[j] = (v[j] - m) * inv * g[j] + be[j];
}

// M=1 cross-attention: q[32] vs K/V [32 tokens x 32] (heads in col halves).
__device__ __forceinline__ void cross_attn32(const h16 *q, const h16 *K, const h16 *V,
                                             h16 *out, int lane) {
  float p0 = 0.f, p1 = 0.f;
#pragma unroll
  for (int hh = 0; hh < 2; ++hh) {
    float s = -1e30f;
    if (lane < T) {
      float a = 0.f;
#pragma unroll
      for (int d = 0; d < 16; ++d)
        a += (float)q[hh * 16 + d] * (float)K[lane * 32 + hh * 16 + d];
      s = a * 0.25f; // 1/sqrt(16)
    }
    float mx = s;
#pragma unroll
    for (int off = 16; off > 0; off >>= 1) mx = fmaxf(mx, __shfl_xor(mx, off, 32));
    float e = (lane < T) ? __expf(s - mx) : 0.f;
    float den = e;
#pragma unroll
    for (int off = 16; off > 0; off >>= 1) den += __shfl_xor(den, off, 32);
    float pv = e / den;
    if (hh == 0) p0 = pv; else p1 = pv;
  }
  const int hh = lane >> 4, d = lane & 15;
  float acc = 0.f;
#pragma unroll
  for (int j = 0; j < T; ++j) {
    float pj0 = __shfl(p0, j, 32), pj1 = __shfl(p1, j, 32);
    float pj = hh ? pj1 : pj0;
    acc += pj * (float)V[j * 32 + hh * 16 + d];
  }
  out[lane] = (h16)acc;
}

__global__ void __launch_bounds__(128, 1) trm_kernel(TrmParams p) {
  extern __shared__ char smem[];
  const int tid = threadIdx.x, lane = tid & 31, wv = tid >> 5;
  const int b0 = blockIdx.x * HB;

  h16 *W_SAIN = (h16 *)(smem + O_SAIN),  *W_SAOUT = (h16 *)(smem + O_SAOUT);
  h16 *W_FF1  = (h16 *)(smem + O_FF1),   *W_FF2   = (h16 *)(smem + O_FF2);
  h16 *W_CE0Q = (h16 *)(smem + O_CE0Q),  *W_CE0O  = (h16 *)(smem + O_CE0O);
  h16 *W_CETI = (h16 *)(smem + O_CETIN), *W_CETO  = (h16 *)(smem + O_CETO);
  h16 *W_ZFF1 = (h16 *)(smem + O_ZFF1),  *W_ZFF2  = (h16 *)(smem + O_ZFF2);
  h16 *W_YFF1 = (h16 *)(smem + O_YFF1),  *W_YFF2  = (h16 *)(smem + O_YFF2);
  h16 *Ets = (h16 *)(smem + O_ET);
  h16 *K0  = (h16 *)(smem + O_K0), *V0 = (h16 *)(smem + O_V0);
  h16 *ZH  = (h16 *)(smem + O_ZH);
  float *Z = (float *)(smem + O_Z), *Y = (float *)(smem + O_Y);
  h16 *Q0h = (h16 *)(smem + O_Q0), *QTh = (h16 *)(smem + O_QT);
  h16 *AO0 = (h16 *)(smem + O_AO0), *AOT = (h16 *)(smem + O_AOT);
  h16 *ZCAT = (h16 *)(smem + O_ZCAT), *YCAT = (h16 *)(smem + O_YCAT);
  float *BHf = (float *)(smem + O_BH);
  float *BQ0 = BHf, *BQT = BHf + HB * 32;
  h16 *BH16 = (h16 *)(smem + O_BH16);
  float *SCF = (float *)(smem + O_PW + wv * PW_SZ + PW_SCF);
  h16 *SATT  = (h16 *)(smem + O_PW + wv * PW_SZ + PW_ATT);
  // attention-phase carve (halfs): SQ0 0 | SK0 1024 | SQ1 2048 | SK1 3072 |
  // SVT0 4096 | SVT1 4608 | SPR0 5120 | SPR1 6144 | SAO 7168
  // ff/cet-phase carve: SH 0(2048) | SKT 2048(1024) | SVTc 3072(1024)

  if (tid < HB) __builtin_prefetch(p.x + (size_t)(b0 + tid) * XW, 0, 3);

  // ---------------- weights -> LDS f16 ----------------
#define CPW(dst, src, n)                                                               \
  for (int i = tid; i < (n); i += 128) (dst)[i] = (h16)(src)[i];
  CPW(W_SAIN, p.sa_in_w, 96 * 32) CPW(W_SAOUT, p.sa_out_w, 1024)
  CPW(W_FF1, p.sa_ff_w1, 2048)    CPW(W_FF2, p.sa_ff_w2, 2048)
  CPW(W_CE0Q, p.ce0_in_w, 1024)   CPW(W_CE0O, p.ce0_out_w, 1024)
  CPW(W_CETI, p.cet_in_w, 3072)   CPW(W_CETO, p.cet_out_w, 1024)
  CPW(W_ZFF1, p.zff_w1, 6144)     CPW(W_ZFF2, p.zff_w2, 2048)
  CPW(W_YFF1, p.yff_w1, 4096)     CPW(W_YFF2, p.yff_w2, 2048)
  h16 *kvtmp = (h16 *)BHf; // prologue temp: ce0_in_w rows 32..95 as f16
  for (int i = tid; i < 64 * 32; i += 128) kvtmp[i] = (h16)p.ce0_in_w[1024 + i];
#undef CPW

  // zero Et (pad rows stay zero forever)
  for (int i = tid; i < HB * 1024; i += 128) Ets[i] = (h16)0.f;
  __syncthreads();

  // ---------------- prologue: E0 -> Et, z0, y0 ----------------
  for (int r = tid; r < HB * T; r += 128) {
    const int bb = r / T, t = r - bb * T;
    const float *xr = p.x + (size_t)(b0 + bb) * XW + t * 6;
    float v[32];
#pragma unroll
    for (int j = 0; j < 32; ++j) {
      float s = p.tp_b[j];
#pragma unroll
      for (int k = 0; k < 6; ++k) s += xr[k] * p.tp_w[j * 6 + k];
      v[j] = s;
    }
    ln32(v, p.tp_g, p.tp_be);
#pragma unroll
    for (int j = 0; j < 32; ++j) Ets[bb * 1024 + t * 32 + j] = (h16)geluf(v[j]);
  }
  if (tid < HB) {
    const float *xr = p.x + (size_t)(b0 + tid) * XW + 132;
    float v[32];
#pragma unroll
    for (int j = 0; j < 32; ++j) {
      float s = p.zi_b[j];
      for (int k = 0; k < 200; ++k) s += xr[k] * p.zi_w[j * 200 + k];
      v[j] = s;
    }
    ln32(v, p.zi_g, p.zi_be);
#pragma unroll
    for (int j = 0; j < 32; ++j) {
      float zv = geluf(v[j]);
      Z[tid * 32 + j] = zv; ZH[tid * 32 + j] = (h16)zv; Y[tid * 32 + j] = 0.f;
    }
  }
  __syncthreads();

  // ---------------- precompute ce0 K/V (E0 is step-invariant) ----------------
  for (int bb = wv; bb < HB; bb += 4) {
    mm_lds<2, 4, 1>(Ets + bb * 1024, 32, kvtmp, 32, SCF, 64, lane);
    for (int i = lane; i < T * 64; i += 32) {
      const int t = i >> 6, f = i & 63;
      float val = SCF[i] + p.ce0_in_b[32 + f];
      if (f < 32) K0[bb * 1024 + t * 32 + f] = (h16)val;
      else        V0[bb * 1024 + t * 32 + (f - 32)] = (h16)val;
    }
  }
  __syncthreads();

  // ================= recurrence =================
  for (int step = 0; step < STEPS; ++step) {
    // ---- self-attention + FF, per wave / per batch element ----
    for (int bb = wv; bb < HB; bb += 4) {
      h16 *EtB = Ets + bb * 1024;
      // qkv: (32pad x 32) @ (96 x 32)^T
      mm_lds<2, 6, 1>(EtB, 32, W_SAIN, 32, SCF, 96, lane);
      for (int i = lane; i < 2048; i += 32) { // Q,K staged zero-padded to K=32
        const int hh = i >> 10, r = (i >> 5) & 31, c = i & 31;
        float qv = 0.f, kv = 0.f;
        if (c < 16) {
          qv = SCF[r * 96 + hh * 16 + c] + p.sa_in_b[hh * 16 + c];
          kv = SCF[r * 96 + 32 + hh * 16 + c] + p.sa_in_b[32 + hh * 16 + c];
        }
        (SATT + hh * 2048)[r * 32 + c] = (h16)qv;
        (SATT + 1024 + hh * 2048)[r * 32 + c] = (h16)kv;
      }
      for (int i = lane; i < 1024; i += 32) { // V^T (hd x tokens)
        const int hh = i >> 9, d = (i >> 5) & 15, m = i & 31;
        (SATT + 4096 + hh * 512)[d * 32 + m] =
            (h16)(SCF[m * 96 + 64 + hh * 16 + d] + p.sa_in_b[64 + hh * 16 + d]);
      }
      __builtin_amdgcn_wave_barrier();
      for (int hh = 0; hh < 2; ++hh) {
        // scores
        mm_lds<2, 2, 1>(SATT + hh * 2048, 32, SATT + 1024 + hh * 2048, 32, SCF, 32, lane);
        { // softmax row = lane, mask pad keys
          h16 *pr = SATT + 5120 + hh * 1024 + lane * 32;
          if (lane < T) {
            float mx = -1e30f, ex[T], den = 0.f;
#pragma unroll
            for (int j = 0; j < T; ++j) mx = fmaxf(mx, SCF[lane * 32 + j] * 0.25f);
#pragma unroll
            for (int j = 0; j < T; ++j) {
              ex[j] = __expf(SCF[lane * 32 + j] * 0.25f - mx); den += ex[j];
            }
            float inv = 1.f / den;
#pragma unroll
            for (int j = 0; j < T; ++j) pr[j] = (h16)(ex[j] * inv);
#pragma unroll
            for (int j = T; j < 32; ++j) pr[j] = (h16)0.f;
          } else {
#pragma unroll
            for (int j = 0; j < 32; ++j) pr[j] = (h16)0.f;
          }
        }
        __builtin_amdgcn_wave_barrier();
        // probs @ V  -> f32 attn-out (tokens x 16) at col hh*16
        mm_lds<2, 1, 1>(SATT + 5120 + hh * 1024, 32, SATT + 4096 + hh * 512, 32,
                        SCF + 1024 + hh * 16, 32, lane);
      }
      for (int i = lane; i < 1024; i += 32) (SATT + 7168)[i] = (h16)SCF[1024 + i];
      __builtin_amdgcn_wave_barrier();
      // out-proj + residual + LN
      mm_lds<2, 2, 1>(SATT + 7168, 32, W_SAOUT, 32, SCF, 32, lane);
      if (lane < T) {
        float v[32];
#pragma unroll
        for (int j = 0; j < 32; ++j)
          v[j] = (float)EtB[lane * 32 + j] + SCF[lane * 32 + j] + p.sa_out_b[j];
        ln32(v, p.sa_norm_g, p.sa_norm_b);
#pragma unroll
        for (int j = 0; j < 32; ++j) EtB[lane * 32 + j] = (h16)v[j];
      }
      __builtin_amdgcn_wave_barrier();
      // FF
      mm_lds<2, 4, 1>(EtB, 32, W_FF1, 32, SCF, 64, lane);
      for (int i = lane; i < 2048; i += 32)
        SATT[i] = (h16)geluf(SCF[i] + p.sa_ff_b1[i & 63]); // SH
      __builtin_amdgcn_wave_barrier();
      mm_lds<2, 2, 2>(SATT, 64, W_FF2, 64, SCF, 32, lane);
      if (lane < T) {
        float v[32];
#pragma unroll
        for (int j = 0; j < 32; ++j)
          v[j] = (float)EtB[lane * 32 + j] + SCF[lane * 32 + j] + p.sa_ff_b2[j];
        ln32(v, p.sa_ffn_g, p.sa_ffn_b);
#pragma unroll
        for (int j = 0; j < 32; ++j) EtB[lane * 32 + j] = (h16)v[j];
      }
      __builtin_amdgcn_wave_barrier();
    }
    __syncthreads();

    // ---- z path: batched q projections (M = 16 batch rows) ----
    if (wv < 2)
      mm_lds<1, 1, 1>(ZH, 32, W_CE0Q + wv * 16 * 32, 32, BQ0 + wv * 16, 32, lane);
    else
      mm_lds<1, 1, 1>(ZH, 32, W_CETI + (wv - 2) * 16 * 32, 32, BQT + (wv - 2) * 16, 32, lane);
    __syncthreads();
    for (int i = tid; i < HB * 32; i += 128) {
      Q0h[i] = (h16)(BQ0[i] + p.ce0_in_b[i & 31]);
      QTh[i] = (h16)(BQT[i] + p.cet_in_b[i & 31]);
    }
    __syncthreads();

    // ---- per wave: cet k/v + both cross-attentions ----
    for (int bb = wv; bb < HB; bb += 4) {
      mm_lds<2, 4, 1>(Ets + bb * 1024, 32, W_CETI + 32 * 32, 32, SCF, 64, lane);
      for (int i = lane; i < T * 64; i += 32) {
        const int t = i >> 6, f = i & 63;
        float val = SCF[i] + p.cet_in_b[32 + f];
        if (f < 32) (SATT + 2048)[t * 32 + f] = (h16)val;       // SKT
        else        (SATT + 3072)[t * 32 + (f - 32)] = (h16)val; // SVTc
      }
      __builtin_amdgcn_wave_barrier();
      cross_attn32(Q0h + bb * 32, K0 + bb * 1024, V0 + bb * 1024, AO0 + bb * 32, lane);
      cross_attn32(QTh + bb * 32, SATT + 2048, SATT + 3072, AOT + bb * 32, lane);
    }
    __syncthreads();

    // ---- z out-projections (batched) ----
    if (wv < 2)
      mm_lds<1, 1, 1>(AO0, 32, W_CE0O + wv * 16 * 32, 32, BQ0 + wv * 16, 32, lane);
    else
      mm_lds<1, 1, 1>(AOT, 32, W_CETO + (wv - 2) * 16 * 32, 32, BQT + (wv - 2) * 16, 32, lane);
    __syncthreads();
    if (tid < HB) {
      float ve[32], vt[32];
#pragma unroll
      for (int j = 0; j < 32; ++j) {
        float zj = Z[tid * 32 + j];
        ve[j] = zj + BQ0[tid * 32 + j] + p.ce0_out_b[j];
        vt[j] = zj + BQT[tid * 32 + j] + p.cet_out_b[j];
      }
      ln32(ve, p.ce0_norm_g, p.ce0_norm_b);
      ln32(vt, p.cet_norm_g, p.cet_norm_b);
#pragma unroll
      for (int j = 0; j < 32; ++j) {
        ZCAT[tid * 96 + j]      = ZH[tid * 32 + j];
        ZCAT[tid * 96 + 32 + j] = (h16)ve[j];
        ZCAT[tid * 96 + 64 + j] = (h16)vt[j];
      }
    }
    __syncthreads();

    // ---- zff: (16x96)->(64)->gelu->(32), LN, z += ----
    mm_lds<1, 1, 3>(ZCAT, 96, W_ZFF1 + wv * 16 * 96, 96, BHf + wv * 16, 64, lane);
    __syncthreads();
    for (int i = tid; i < HB * 64; i += 128)
      BH16[i] = (h16)geluf(BHf[i] + p.zff_b1[i & 63]);
    __syncthreads();
    if (wv < 2)
      mm_lds<1, 1, 2>(BH16, 64, W_ZFF2 + wv * 16 * 64, 64, BQ0 + wv * 16, 32, lane);
    __syncthreads();
    if (tid < HB) {
      float v[32];
#pragma unroll
      for (int j = 0; j < 32; ++j) v[j] = BQ0[tid * 32 + j] + p.zff_b2[j];
      ln32(v, p.zff_g, p.zff_be);
#pragma unroll
      for (int j = 0; j < 32; ++j) {
        float zn = Z[tid * 32 + j] + v[j];
        Z[tid * 32 + j] = zn;
        ZH[tid * 32 + j] = (h16)zn;
        YCAT[tid * 64 + j]      = (h16)Y[tid * 32 + j]; // old y
        YCAT[tid * 64 + 32 + j] = (h16)zn;              // new z
      }
    }
    __syncthreads();

    // ---- yff: (16x64)->(64)->gelu->(32), LN, y += ----
    mm_lds<1, 1, 2>(YCAT, 64, W_YFF1 + wv * 16 * 64, 64, BHf + wv * 16, 64, lane);
    __syncthreads();
    for (int i = tid; i < HB * 64; i += 128)
      BH16[i] = (h16)geluf(BHf[i] + p.yff_b1[i & 63]);
    __syncthreads();
    if (wv < 2)
      mm_lds<1, 1, 2>(BH16, 64, W_YFF2 + wv * 16 * 64, 64, BQ0 + wv * 16, 32, lane);
    __syncthreads();
    if (tid < HB) {
      float v[32];
#pragma unroll
      for (int j = 0; j < 32; ++j) v[j] = BQ0[tid * 32 + j] + p.yff_b2[j];
      ln32(v, p.yff_g, p.yff_be);
#pragma unroll
      for (int j = 0; j < 32; ++j) Y[tid * 32 + j] += v[j];
    }
    __syncthreads();
  }

  // ---------------- head ----------------
  if (tid < HB) {
    float s = p.head_b[0];
#pragma unroll
    for (int j = 0; j < 32; ++j) s += Y[tid * 32 + j] * p.head_w[j];
    p.out[b0 + tid] = s;
  }
}

} // namespace trm

extern "C" void kernel_launch(void *const *d_in, const int *in_sizes, int n_in,
                              void *d_out, int out_size, void *d_ws, size_t ws_size,
                              hipStream_t stream) {
  (void)in_sizes; (void)n_in; (void)out_size; (void)d_ws; (void)ws_size;
  trm::TrmParams P;
  const float **f = (const float **)d_in;
  P.x = f[0];
  P.tp_w = f[1];  P.tp_b = f[2];  P.tp_g = f[3];  P.tp_be = f[4];
  P.zi_w = f[5];  P.zi_b = f[6];  P.zi_g = f[7];  P.zi_be = f[8];
  P.sa_in_w = f[9];  P.sa_in_b = f[10]; P.sa_out_w = f[11]; P.sa_out_b = f[12];
  P.sa_norm_g = f[13]; P.sa_norm_b = f[14];
  P.sa_ff_w1 = f[15]; P.sa_ff_b1 = f[16]; P.sa_ff_w2 = f[17]; P.sa_ff_b2 = f[18];
  P.sa_ffn_g = f[19]; P.sa_ffn_b = f[20];
  P.ce0_in_w = f[21]; P.ce0_in_b = f[22]; P.ce0_out_w = f[23]; P.ce0_out_b = f[24];
  P.ce0_norm_g = f[25]; P.ce0_norm_b = f[26];
  P.cet_in_w = f[27]; P.cet_in_b = f[28]; P.cet_out_w = f[29]; P.cet_out_b = f[30];
  P.cet_norm_g = f[31]; P.cet_norm_b = f[32];
  P.zff_w1 = f[33]; P.zff_b1 = f[34]; P.zff_w2 = f[35]; P.zff_b2 = f[36];
  P.zff_g = f[37]; P.zff_be = f[38];
  P.yff_w1 = f[39]; P.yff_b1 = f[40]; P.yff_w2 = f[41]; P.yff_b2 = f[42];
  P.yff_g = f[43]; P.yff_be = f[44];
  P.head_w = f[45]; P.head_b = f[46];
  P.out = (float *)d_out;

  dim3 grid(trm::Bsz / trm::HB), block(128);
  trm::trm_kernel<<<grid, block, trm::LDS_BYTES, stream>>>(P);
}
